// PTBlock_38019050504226
// MI455X (gfx1250) — matrix operand order repaired and checked by reading, not compiled, not run on hardware
//
#include <hip/hip_runtime.h>

// ---------------------------------------------------------------------------
// Point-Transformer block, fused for MI455X (gfx1250, wave32, bf16 WMMA).
//   B=4, C=256, N=4096, S=16, H=128
// All GEMM stages use v_wmma_f32_16x16x32_bf16. One wave owns one point (b,n)
// in the main kernel; its S=16 samples are the M dim of every WMMA tile so the
// softmax over S is an in-tile column reduction (8 regs + shfl_xor 16).
// ---------------------------------------------------------------------------

#define Bdim 4
#define Cdim 256
#define Ndim 4096
#define Sdim 16
#define Hdim 128

typedef __attribute__((ext_vector_type(8)))  float  v8f;
typedef __attribute__((ext_vector_type(16))) __bf16 v16bf;

union AFrag {
  v16bf v;
  unsigned short u[16];
  uint4 q[2];
};

// swizzled-weight tile bases (each tile = 512 bf16 = one B fragment set)
enum {
  BT_top   = 0,    // Cin=256, O=128 : 8k x 8h
  BT_phi   = 64,   // 128->128      : 4k x 8h
  BT_psi   = 96,   // 256->128
  BT_alpha = 160,
  BT_d1    = 224,  // 3(pad 32)->128: 1k x 8h
  BT_d2    = 232,  // 128->128
  BT_g1    = 264,
  BT_g2    = 296,
  BT_down  = 328,  // 128->256      : 4k x 16h
  BT_total = 392
};

__device__ __forceinline__ unsigned short f2bf(float x) {
  unsigned int u = __float_as_uint(x);
  u = (u + 0x7FFFu + ((u >> 16) & 1u)) >> 16;   // round-to-nearest-even
  return (unsigned short)u;
}

__device__ __forceinline__ v8f wmma_bf16(v16bf a, v16bf b, v8f c) {
  return __builtin_amdgcn_wmma_f32_16x16x32_bf16(false, a, false, b,
                                                 (short)0, c, false, false);
}

// B fragment: 16 contiguous bf16 per lane (two global_load_b128)
__device__ __forceinline__ v16bf load_bfrag(const unsigned short* __restrict__ swz,
                                            int tile, int lane) {
  AFrag f;
  const uint4* p = (const uint4*)(swz + (size_t)tile * 512 + lane * 16);
  f.q[0] = p[0];
  f.q[1] = p[1];
  return f.v;
}

// A fragment from a row-major bf16 LDS tile: two ds_load_b128 per lane.
// row points at &lds[s*width]; c0 = ktile*32 + 8*half.
__device__ __forceinline__ v16bf load_afrag_lds(const unsigned short* row, int c0) {
  AFrag f;
  f.q[0] = *(const uint4*)(row + c0);
  f.q[1] = *(const uint4*)(row + c0 + 16);
  return f.v;
}

#define V8F_ZERO {0.f,0.f,0.f,0.f,0.f,0.f,0.f,0.f}

// ---------------------------------------------------------------------------
// Kernel 0: swizzle all weights into bf16 B-fragment layout (once per launch).
// One 32-lane block per 32x16 tile. Element i of lane L = W[n=L%16][k=8*(L/16)+i+(i>=8?8:0)].
// ---------------------------------------------------------------------------
__global__ void pt_prep(const float* __restrict__ Wtop,  const float* __restrict__ Wphi,
                        const float* __restrict__ Wpsi,  const float* __restrict__ Walpha,
                        const float* __restrict__ Wd1,   const float* __restrict__ Wd2,
                        const float* __restrict__ Wg1,   const float* __restrict__ Wg2,
                        const float* __restrict__ Wdown, unsigned short* __restrict__ swz) {
  int tile = blockIdx.x;
  int lane = threadIdx.x & 31;
  const float* W; int base, ktiles, cin;
  if      (tile < BT_phi)   { W = Wtop;   base = BT_top;   ktiles = 8; cin = 256; }
  else if (tile < BT_psi)   { W = Wphi;   base = BT_phi;   ktiles = 4; cin = 128; }
  else if (tile < BT_alpha) { W = Wpsi;   base = BT_psi;   ktiles = 8; cin = 256; }
  else if (tile < BT_d1)    { W = Walpha; base = BT_alpha; ktiles = 8; cin = 256; }
  else if (tile < BT_d2)    { W = Wd1;    base = BT_d1;    ktiles = 1; cin = 3;   }
  else if (tile < BT_g1)    { W = Wd2;    base = BT_d2;    ktiles = 4; cin = 128; }
  else if (tile < BT_g2)    { W = Wg1;    base = BT_g1;    ktiles = 4; cin = 128; }
  else if (tile < BT_down)  { W = Wg2;    base = BT_g2;    ktiles = 4; cin = 128; }
  else                      { W = Wdown;  base = BT_down;  ktiles = 4; cin = 128; }
  int rel   = tile - base;
  int ktile = rel % ktiles;
  int htile = rel / ktiles;
  int half  = lane >> 4, lcol = lane & 15;
  unsigned short* dst = swz + (size_t)tile * 512 + lane * 16;
#pragma unroll
  for (int i = 0; i < 16; ++i) {
    int k = ktile * 32 + 8 * half + i + (i >= 8 ? 8 : 0);
    float v = (k < cin) ? W[(htile * 16 + lcol) * cin + k] : 0.0f;
    dst[i] = f2bf(v);
  }
}

// ---------------------------------------------------------------------------
// Kernel 1: phi_ws[token,h] = W_phi @ (W_top @ x_cent + b_top) + b_phi  (f32)
// One wave per 16 consecutive tokens (same b since N%16==0).
// ---------------------------------------------------------------------------
__global__ void pt_top_phi(const float* __restrict__ xc_in,   // [B,C,N]
                           const unsigned short* __restrict__ swz,
                           const float* __restrict__ btop, const float* __restrict__ bphi,
                           float* __restrict__ phi_ws) {     // [B*N,128]
  extern __shared__ unsigned short smem1[];
  int wave = threadIdx.x >> 5, lane = threadIdx.x & 31;
  int half = lane >> 4, lcol = lane & 15;
  int grp  = blockIdx.x * 4 + wave;
  int tok0 = grp * 16;
  int b    = tok0 >> 12;
  int n0   = tok0 & (Ndim - 1);
  unsigned short* lds_xi = smem1 + (size_t)wave * 6144;        // [16][256] bf16
  unsigned short* lds_xc = lds_xi + 4096;                      // [16][128] bf16

  // stage x_cent tile [16 tokens][256 ch] row-major (token rows)
  for (int c = lane; c < Cdim; c += 32) {
    const float* src = xc_in + ((size_t)(b * Cdim) + c) * Ndim + n0;
#pragma unroll
    for (int s = 0; s < 16; ++s) lds_xi[s * Cdim + c] = f2bf(src[s]);
  }

  // xc = W_top @ x + b_top   -> lds_xc (bf16)
  for (int t = 0; t < 8; ++t) {
    v8f acc = V8F_ZERO;
#pragma unroll
    for (int kt = 0; kt < 8; ++kt) {
      v16bf a = load_afrag_lds(lds_xi + lcol * Cdim, kt * 32 + 8 * half);
      acc = wmma_bf16(a, load_bfrag(swz, BT_top + t * 8 + kt, lane), acc);
    }
    float bias = btop[t * 16 + lcol];
#pragma unroll
    for (int r = 0; r < 8; ++r)
      lds_xc[(r + 8 * half) * Hdim + t * 16 + lcol] = f2bf(acc[r] + bias);
  }

  // phi = W_phi @ xc + b_phi -> phi_ws (f32)
  for (int t = 0; t < 8; ++t) {
    v8f acc = V8F_ZERO;
#pragma unroll
    for (int kt = 0; kt < 4; ++kt) {
      v16bf a = load_afrag_lds(lds_xc + lcol * Hdim, kt * 32 + 8 * half);
      acc = wmma_bf16(a, load_bfrag(swz, BT_phi + t * 4 + kt, lane), acc);
    }
    float bias = bphi[t * 16 + lcol];
#pragma unroll
    for (int r = 0; r < 8; ++r)
      phi_ws[(size_t)(tok0 + r + 8 * half) * Hdim + t * 16 + lcol] = acc[r] + bias;
  }
}

// ---------------------------------------------------------------------------
// Kernel 2: fused psi/alpha/delta/g1/g2 + softmax(S) + weighted sum.
// One wave per (b,n). M dim of every WMMA tile = the 16 samples.
// ---------------------------------------------------------------------------
__global__ void pt_main(const float* __restrict__ p_cent,  // [B,3,N]
                        const float* __restrict__ p,       // [B,3,N,S]
                        const float* __restrict__ x,       // [B,C,N,S]
                        const float* __restrict__ phi_ws,  // [B*N,128]
                        const unsigned short* __restrict__ swz,
                        const float* __restrict__ bpsi, const float* __restrict__ balpha,
                        const float* __restrict__ bd1,  const float* __restrict__ bd2,
                        const float* __restrict__ bg1,  const float* __restrict__ bg2,
                        unsigned short* __restrict__ y_ws) { // [B*N,128] bf16
  extern __shared__ unsigned short smem2[];
  const int wave  = threadIdx.x >> 5;
  const int lane  = threadIdx.x & 31;
  const int token = blockIdx.x * 4 + wave;
  const int b     = token >> 12;
  const int n     = token & (Ndim - 1);
  const int half  = lane >> 4, lcol = lane & 15;

  char* base = (char*)smem2 + (size_t)wave * 24576;
  unsigned short* lds_x  = (unsigned short*)base;            // [16][256] bf16 (8KB)
  unsigned short* lds_t  = (unsigned short*)(base + 8192);   // [16][128] bf16 (4KB) relu(d1) then relu(g1u)
  unsigned short* lds_u  = (unsigned short*)(base + 12288);  // [16][128] bf16 (4KB)
  float*          lds_ad = (float*)(base + 16384);           // [16][128] f32  (8KB)

  // ---- phase 0: stage x_j tile: input_x[b,:,n,:] -> lds_x [s][c] bf16 ----
  const float* xg = x + (size_t)(b * Cdim) * Ndim * Sdim + (size_t)n * Sdim;
  for (int c = lane; c < Cdim; c += 32) {
    const float* src = xg + (size_t)c * Ndim * Sdim;
#pragma unroll
    for (int s = 0; s < 16; ++s) lds_x[s * Cdim + c] = f2bf(src[s]);
  }

  // relative positions for this lane's sample row (K=3, zero-padded to 32)
  AFrag apip;
  float pipv[3];
#pragma unroll
  for (int k = 0; k < 3; ++k) {
    float pc = p_cent[((size_t)b * 3 + k) * Ndim + n];
    float pv = p[(((size_t)b * 3 + k) * Ndim + n) * Sdim + lcol];
    pipv[k] = pc - pv;
  }
#pragma unroll
  for (int i = 0; i < 16; ++i) {
    int kl = i + 8 * half + (i >= 8 ? 8 : 0);
    apip.u[i] = (kl < 3) ? f2bf(pipv[kl]) : (unsigned short)0;
  }

  // ---- phase 1: t1 = relu(W_d1 @ pipj + b_d1) -> lds_t ----
  for (int t = 0; t < 8; ++t) {
    v8f acc = V8F_ZERO;
    acc = wmma_bf16(apip.v, load_bfrag(swz, BT_d1 + t, lane), acc);
    float bias = bd1[t * 16 + lcol];
#pragma unroll
    for (int r = 0; r < 8; ++r) {
      float v = acc[r] + bias;
      v = v > 0.f ? v : 0.f;
      lds_t[(r + 8 * half) * Hdim + t * 16 + lcol] = f2bf(v);
    }
  }

  // ---- phase 2: delta, psi, alpha; u = phi - psi + delta; ad = alpha+delta ----
  for (int t = 0; t < 8; ++t) {
    v8f accd = V8F_ZERO, accp = V8F_ZERO, acca = V8F_ZERO;
#pragma unroll
    for (int kt = 0; kt < 4; ++kt) {
      v16bf a = load_afrag_lds(lds_t + lcol * Hdim, kt * 32 + 8 * half);
      accd = wmma_bf16(a, load_bfrag(swz, BT_d2 + t * 4 + kt, lane), accd);
    }
#pragma unroll
    for (int kt = 0; kt < 8; ++kt) {
      v16bf a = load_afrag_lds(lds_x + lcol * Cdim, kt * 32 + 8 * half);
      accp = wmma_bf16(a, load_bfrag(swz, BT_psi   + t * 8 + kt, lane), accp);
      acca = wmma_bf16(a, load_bfrag(swz, BT_alpha + t * 8 + kt, lane), acca);
    }
    int h = t * 16 + lcol;
    float vbd = bd2[h], vbp = bpsi[h], vba = balpha[h];
    float phiv = phi_ws[(size_t)token * Hdim + h];   // constant over the 16 samples
#pragma unroll
    for (int r = 0; r < 8; ++r) {
      int m = r + 8 * half;
      float d  = accd[r] + vbd;
      float ps = accp[r] + vbp;
      float al = acca[r] + vba;
      lds_u[m * Hdim + h]  = f2bf(phiv - ps + d);
      lds_ad[m * Hdim + h] = al + d;
    }
  }

  // ---- phase 3: v = relu(W_g1 @ u + b_g1) -> lds_t (reused) ----
  for (int t = 0; t < 8; ++t) {
    v8f acc = V8F_ZERO;
#pragma unroll
    for (int kt = 0; kt < 4; ++kt) {
      v16bf a = load_afrag_lds(lds_u + lcol * Hdim, kt * 32 + 8 * half);
      acc = wmma_bf16(a, load_bfrag(swz, BT_g1 + t * 4 + kt, lane), acc);
    }
    float bias = bg1[t * 16 + lcol];
#pragma unroll
    for (int r = 0; r < 8; ++r) {
      float v = acc[r] + bias;
      v = v > 0.f ? v : 0.f;
      lds_t[(r + 8 * half) * Hdim + t * 16 + lcol] = f2bf(v);
    }
  }

  // ---- phase 4: g = W_g2 @ v + b_g2; softmax over S (tile rows); y = sum attn*ad ----
  for (int t = 0; t < 8; ++t) {
    v8f acc = V8F_ZERO;
#pragma unroll
    for (int kt = 0; kt < 4; ++kt) {
      v16bf a = load_afrag_lds(lds_t + lcol * Hdim, kt * 32 + 8 * half);
      acc = wmma_bf16(a, load_bfrag(swz, BT_g2 + t * 4 + kt, lane), acc);
    }
    int h = t * 16 + lcol;
    float bias = bg2[h];
    float g[8], mx = -3.4e38f;
#pragma unroll
    for (int r = 0; r < 8; ++r) { g[r] = acc[r] + bias; mx = fmaxf(mx, g[r]); }
    mx = fmaxf(mx, __shfl_xor(mx, 16, 32));          // combine lane halves (rows 0-7 vs 8-15)
    float sm = 0.f;
#pragma unroll
    for (int r = 0; r < 8; ++r) { g[r] = __expf(g[r] - mx); sm += g[r]; }
    sm += __shfl_xor(sm, 16, 32);
    float inv = 1.0f / sm;
    float part = 0.f;
#pragma unroll
    for (int r = 0; r < 8; ++r) {
      int m = r + 8 * half;
      part += g[r] * inv * lds_ad[m * Hdim + h];
    }
    part += __shfl_xor(part, 16, 32);
    if (half == 0) y_ws[(size_t)token * Hdim + h] = f2bf(part);
  }
}

// ---------------------------------------------------------------------------
// Kernel 3: out[b,o,n] = W_down @ y + b_down + x_cent   (one wave / 16 tokens)
// ---------------------------------------------------------------------------
__global__ void pt_down(const unsigned short* __restrict__ y_ws, // [B*N,128] bf16
                        const unsigned short* __restrict__ swz,
                        const float* __restrict__ bdown,
                        const float* __restrict__ xres,           // [B,C,N]
                        float* __restrict__ out) {                // [B,C,N]
  int wave = threadIdx.x >> 5, lane = threadIdx.x & 31;
  int half = lane >> 4, lcol = lane & 15;
  int grp  = blockIdx.x * 4 + wave;
  int tok0 = grp * 16;
  int b    = tok0 >> 12;
  int n0   = tok0 & (Ndim - 1);

  // A fragments (4 k-tiles) straight from global: y rows are contiguous bf16
  AFrag afr[4];
  const unsigned short* row = y_ws + (size_t)(tok0 + lcol) * Hdim;
#pragma unroll
  for (int kt = 0; kt < 4; ++kt) {
    afr[kt].q[0] = *(const uint4*)(row + kt * 32 + 8 * half);
    afr[kt].q[1] = *(const uint4*)(row + kt * 32 + 8 * half + 16);
  }
  for (int t = 0; t < 16; ++t) {
    v8f acc = V8F_ZERO;
#pragma unroll
    for (int kt = 0; kt < 4; ++kt)
      acc = wmma_bf16(afr[kt].v, load_bfrag(swz, BT_down + t * 4 + kt, lane), acc);
    int o = t * 16 + lcol;
    float bias = bdown[o];
    const float* res = xres + ((size_t)(b * Cdim) + o) * Ndim + n0;
    float*       dst = out  + ((size_t)(b * Cdim) + o) * Ndim + n0;
#pragma unroll
    for (int r = 0; r < 8; ++r) {
      int m = r + 8 * half;
      dst[m] = acc[r] + bias + res[m];
    }
  }
}

// ---------------------------------------------------------------------------
extern "C" void kernel_launch(void* const* d_in, const int* in_sizes, int n_in,
                              void* d_out, int out_size, void* d_ws, size_t ws_size,
                              hipStream_t stream) {
  const float* p_cent = (const float*)d_in[0];
  const float* x_cent = (const float*)d_in[1];
  const float* p      = (const float*)d_in[2];
  const float* x      = (const float*)d_in[3];
  const float* Wtop   = (const float*)d_in[4];  const float* btop   = (const float*)d_in[5];
  const float* Wdown  = (const float*)d_in[6];  const float* bdown  = (const float*)d_in[7];
  const float* Wphi   = (const float*)d_in[8];  const float* bphi   = (const float*)d_in[9];
  const float* Wpsi   = (const float*)d_in[10]; const float* bpsi   = (const float*)d_in[11];
  const float* Walpha = (const float*)d_in[12]; const float* balpha = (const float*)d_in[13];
  const float* Wg1    = (const float*)d_in[14]; const float* bg1    = (const float*)d_in[15];
  const float* Wg2    = (const float*)d_in[16]; const float* bg2    = (const float*)d_in[17];
  const float* Wd1    = (const float*)d_in[18]; const float* bd1    = (const float*)d_in[19];
  const float* Wd2    = (const float*)d_in[20]; const float* bd2    = (const float*)d_in[21];
  float* out = (float*)d_out;

  char* ws = (char*)d_ws;
  unsigned short* swz    = (unsigned short*)ws;                       // 392*512*2 B
  float*          phi_ws = (float*)(ws + 524288);                     // 8 MB
  unsigned short* y_ws   = (unsigned short*)(ws + 524288 + 8388608);  // 4 MB

  pt_prep<<<BT_total, 32, 0, stream>>>(Wtop, Wphi, Wpsi, Walpha, Wd1, Wd2,
                                       Wg1, Wg2, Wdown, swz);
  pt_top_phi<<<(Bdim * Ndim) / 64, 128, 49152, stream>>>(x_cent, swz, btop, bphi, phi_ws);
  pt_main<<<(Bdim * Ndim) / 4, 128, 98304, stream>>>(p_cent, p, x, phi_ws, swz,
                                                     bpsi, balpha, bd1, bd2, bg1, bg2, y_ws);
  pt_down<<<(Bdim * Ndim) / 64, 128, 0, stream>>>(y_ws, swz, bdown, x_cent, out);
}
